// CFLayer_22385369546867
// MI455X (gfx1250) — compile-verified, matrix operand order checked by reading, not run on hardware
//
#include <hip/hip_runtime.h>

// ---------------- problem constants ----------------
#define BATCH 4
#define PCPT_ 1024
#define GEN_  1024
#define SEQ   2048
#define DMODEL 512
#define NHEAD 8
#define HDIM  64
#define DFF_  2048
#define MTOT  (BATCH * SEQ)   // 8192 rows

typedef _Float16 f16;
typedef _Float16 v16h  __attribute__((ext_vector_type(16)));
typedef _Float16 f16x8 __attribute__((ext_vector_type(8)));
typedef float    v8f   __attribute__((ext_vector_type(8)));

// WMMA: D = A(16x32 f16) * B(32x16 f16) + C (fp32 accum)
__device__ __forceinline__ v8f wmma16(v16h a, v16h b, v8f c) {
  return __builtin_amdgcn_wmma_f32_16x16x32_f16(
      false, a, false, b, (short)0, c, false, false);
}

// A/B-layout fragment from a 32-half-strided row segment.
// Per ISA: halves [hb..hb+7] and [hb+16..hb+23], hb = 8*(lane>=16).
__device__ __forceinline__ v16h frag_ld(const f16* p) {
  union { f16x8 h[2]; v16h v; } u;
  u.h[0] = *(const f16x8*)(p);
  u.h[1] = *(const f16x8*)(p + 16);
  return u.v;
}

// 16B-per-lane async DMA into LDS (ASYNCcnt-tracked, bypasses VGPRs).
__device__ __forceinline__ void async_b128(void* lds, const void* g) {
  unsigned loff = (unsigned)(uintptr_t)lds;  // low 32 bits = LDS byte offset
  asm volatile("global_load_async_to_lds_b128 %0, %1, off"
               :: "v"(loff), "v"(g)
               : "memory");
}
__device__ __forceinline__ void wait_async_le4() {
  asm volatile("s_wait_asynccnt 0x4" ::: "memory");
}
__device__ __forceinline__ void wait_async_0() {
  asm volatile("s_wait_asynccnt 0x0" ::: "memory");
}
__device__ __forceinline__ void wait_ds_0() {
  asm volatile("s_wait_dscnt 0x0" ::: "memory");
}

// ---------------- fp32 -> f16 converter (weights, once) ----------------
__global__ __launch_bounds__(256) void k_cvt(const float* __restrict__ src,
                                             f16* __restrict__ dst, int n) {
  int i = blockIdx.x * 256 + threadIdx.x;
  if (i < n) dst[i] = (f16)src[i];
}

// ---------------- concat + fp32->f16 activations ----------------
__global__ __launch_bounds__(256) void k_concat(const float* __restrict__ pcpt,
                                                const float* __restrict__ gen,
                                                f16* __restrict__ xh) {
  int idx = blockIdx.x * 256 + threadIdx.x;       // over MTOT*DMODEL
  int d   = idx & (DMODEL - 1);
  int row = idx >> 9;
  int b   = row >> 11;
  int s   = row & (SEQ - 1);
  float v = (s < PCPT_) ? pcpt[((size_t)b * PCPT_ + s) * DMODEL + d]
                        : gen[((size_t)b * GEN_ + (s - PCPT_)) * DMODEL + d];
  xh[idx] = (f16)v;
}

// ------ WMMA GEMM: C[MxN] = A[MxK](f16) * Wh[NxK]^T(f16) + bias ------------
// 256 thr (8 waves), tile 128x128, k-step 32, double-buffered async LDS fill.
__global__ __launch_bounds__(256) void k_gemm(const f16* __restrict__ A,
                                              const f16* __restrict__ Wh,
                                              const float* __restrict__ bias,
                                              float* __restrict__ of32,
                                              f16* __restrict__ of16,
                                              int M, int N, int K, int relu) {
  __shared__ f16 As[2][128 * 32];
  __shared__ f16 Ws[2][128 * 32];
  const int tid  = threadIdx.x;
  const int lane = tid & 31, wid = tid >> 5;
  const int half = lane >> 4, ln = lane & 15;
  const int m0 = blockIdx.y * 128, n0 = blockIdx.x * 128;
  const int wm = (wid >> 1) * 32, wn = (wid & 1) * 64;

  // fixed per-thread staging chunks: 512 x 16B per tile, 2 per thread
  const int r0 = tid >> 2,          o0 = (tid & 3) * 8;
  const int r1 = (tid + 256) >> 2,  o1 = (tid & 3) * 8;  // (tid+256)&3 == tid&3

  auto stage = [&](int bi, int k0) {
    async_b128(&As[bi][r0 * 32 + o0], &A[(size_t)(m0 + r0) * K + k0 + o0]);
    async_b128(&Ws[bi][r0 * 32 + o0], &Wh[(size_t)(n0 + r0) * K + k0 + o0]);
    async_b128(&As[bi][r1 * 32 + o1], &A[(size_t)(m0 + r1) * K + k0 + o1]);
    async_b128(&Ws[bi][r1 * 32 + o1], &Wh[(size_t)(n0 + r1) * K + k0 + o1]);
  };

  v8f acc[2][4];
  v8f zv = {0.f, 0.f, 0.f, 0.f, 0.f, 0.f, 0.f, 0.f};
#pragma unroll
  for (int mt = 0; mt < 2; ++mt)
#pragma unroll
    for (int nt = 0; nt < 4; ++nt) acc[mt][nt] = zv;

  stage(0, 0);
  stage(1, 32);                          // K >= 64 for all our GEMMs
  int buf = 0;
  for (int k0 = 0; k0 < K; k0 += 32) {
    if (k0 + 32 < K) wait_async_le4();   // batch for `buf` complete (in-order)
    else             wait_async_0();
    __syncthreads();                     // publish DMA'd tile to all waves

    v16h af[2], bf[4];
#pragma unroll
    for (int mt = 0; mt < 2; ++mt)
      af[mt] = frag_ld(&As[buf][(wm + mt * 16 + ln) * 32 + half * 8]);
#pragma unroll
    for (int nt = 0; nt < 4; ++nt)
      bf[nt] = frag_ld(&Ws[buf][(wn + nt * 16 + ln) * 32 + half * 8]);
    wait_ds_0();                         // my fragment reads landed in VGPRs
    __syncthreads();                     // everyone done reading `buf`
    if (k0 + 64 < K) stage(buf, k0 + 64);

#pragma unroll
    for (int mt = 0; mt < 2; ++mt)
#pragma unroll
      for (int nt = 0; nt < 4; ++nt)
        acc[mt][nt] = wmma16(af[mt], bf[nt], acc[mt][nt]);
    buf ^= 1;
  }

  // epilogue: C layout VGPR r -> row r + 8*half, col = ln
#pragma unroll
  for (int mt = 0; mt < 2; ++mt)
#pragma unroll
    for (int nt = 0; nt < 4; ++nt)
#pragma unroll
      for (int r = 0; r < 8; ++r) {
        int gr = m0 + wm + mt * 16 + r + half * 8;
        int gc = n0 + wn + nt * 16 + ln;
        float v = acc[mt][nt][r] + bias[gc];
        if (relu) v = fmaxf(v, 0.f);
        if (of32) of32[(size_t)gr * N + gc] = v;
        if (of16) of16[(size_t)gr * N + gc] = (f16)v;
      }
}

// ---------------- flash attention over the 1024 pcpt keys ----------------
// grid = (SEQ/128, NHEAD, BATCH); 8 waves; wave = 16 queries.
// Writes unnormalized o~ (B,H,S,HD), rowmax m, rowsum l.
__global__ __launch_bounds__(256) void k_flash(const f16* __restrict__ qkv,
                                               const unsigned char* __restrict__ pcpt_kpm,
                                               float* __restrict__ oacc,
                                               float* __restrict__ mrow,
                                               float* __restrict__ lrow) {
  __shared__ f16 Ks[32 * 64];        // 32 keys x 64 dims (async DMA)
  __shared__ f16 Vt[64 * 32];        // transposed: dim-major x 32 keys
  __shared__ f16 Pw[8 * 16 * 32];    // per-wave p relayout scratch
  const int tid = threadIdx.x, lane = tid & 31, wid = tid >> 5;
  const int half = lane >> 4, ln = lane & 15;
  const int b = blockIdx.z, h = blockIdx.y;
  const int q0 = blockIdx.x * 128 + wid * 16;

  // q fragments (16 queries x 64 HD), 2 K-tiles of 32
  v16h aq[2];
  const f16* qbase = qkv + ((size_t)(b * SEQ + q0 + ln)) * (3 * DMODEL) + h * HDIM;
#pragma unroll
  for (int t = 0; t < 2; ++t) {
    union { f16x8 hh[2]; v16h v; } u;
    u.hh[0] = *(const f16x8*)(qbase + t * 32 + half * 8);
    u.hh[1] = *(const f16x8*)(qbase + t * 32 + half * 8 + 16);
    aq[t] = u.v;
  }

  v8f zv = {0.f, 0.f, 0.f, 0.f, 0.f, 0.f, 0.f, 0.f};
  v8f o[4];
#pragma unroll
  for (int nt = 0; nt < 4; ++nt) o[nt] = zv;
  float m_r[8], l_r[8];
#pragma unroll
  for (int r = 0; r < 8; ++r) { m_r[r] = -1e30f; l_r[r] = 0.f; }
  const float scale = 0.125f;  // 1/sqrt(64)
  const int rr = tid >> 3, d0 = (tid & 7) * 8;

  for (int kt = 0; kt < PCPT_; kt += 32) {
    __syncthreads();  // previous iteration fully consumed Ks/Vt
    {
      const f16* kp = qkv + ((size_t)(b * SEQ + kt + rr)) * (3 * DMODEL)
                      + DMODEL + h * HDIM + d0;
      async_b128(&Ks[rr * 64 + d0], kp);           // K tile: DMA straight to LDS
      f16x8 v8 = *(const f16x8*)(kp + DMODEL);     // V tile: VGPR (transpose)
#pragma unroll
      for (int j = 0; j < 8; ++j) Vt[(d0 + j) * 32 + rr] = v8[j];
    }
    wait_async_0();
    __syncthreads();

    // scores for two 16-key column tiles
    v8f c0 = zv, c1 = zv;
#pragma unroll
    for (int t = 0; t < 2; ++t) {
      c0 = wmma16(aq[t], frag_ld(&Ks[(ln) * 64 + t * 32 + half * 8]), c0);
      c1 = wmma16(aq[t], frag_ld(&Ks[(16 + ln) * 64 + t * 32 + half * 8]), c1);
    }
    int km0 = pcpt_kpm[(size_t)b * PCPT_ + kt + ln];
    int km1 = pcpt_kpm[(size_t)b * PCPT_ + kt + 16 + ln];

    f16* pw = &Pw[wid * 16 * 32];
#pragma unroll
    for (int r = 0; r < 8; ++r) {
      float s0 = km0 ? -1e30f : c0[r] * scale;
      float s1 = km1 ? -1e30f : c1[r] * scale;
      float t = fmaxf(s0, s1);
      t = fmaxf(t, __shfl_xor(t, 1, 32));
      t = fmaxf(t, __shfl_xor(t, 2, 32));
      t = fmaxf(t, __shfl_xor(t, 4, 32));
      t = fmaxf(t, __shfl_xor(t, 8, 32));
      float mn = fmaxf(m_r[r], t);
      float al = __expf(m_r[r] - mn);
      float p0 = __expf(s0 - mn);
      float p1 = __expf(s1 - mn);
      float rs = p0 + p1;
      rs += __shfl_xor(rs, 1, 32);
      rs += __shfl_xor(rs, 2, 32);
      rs += __shfl_xor(rs, 4, 32);
      rs += __shfl_xor(rs, 8, 32);
      l_r[r] = l_r[r] * al + rs;
      m_r[r] = mn;
#pragma unroll
      for (int nt = 0; nt < 4; ++nt) o[nt][r] *= al;
      int row = r + half * 8;               // C-layout -> A-layout relayout
      pw[row * 32 + ln]      = (f16)p0;
      pw[row * 32 + 16 + ln] = (f16)p1;
    }
    // p (16x32) as A fragment; V^T columns as B fragments; o += p*V
    v16h ap = frag_ld(&pw[ln * 32 + half * 8]);
#pragma unroll
    for (int nt = 0; nt < 4; ++nt)
      o[nt] = wmma16(ap, frag_ld(&Vt[(nt * 16 + ln) * 32 + half * 8]), o[nt]);
  }

  size_t bh = (size_t)(b * NHEAD + h);
#pragma unroll
  for (int nt = 0; nt < 4; ++nt)
#pragma unroll
    for (int r = 0; r < 8; ++r) {
      int sq = q0 + r + half * 8;
      oacc[(bh * SEQ + sq) * HDIM + nt * 16 + ln] = o[nt][r];
    }
  if (ln == 0)
#pragma unroll
    for (int r = 0; r < 8; ++r) {
      int sq = q0 + r + half * 8;
      mrow[bh * SEQ + sq] = m_r[r];
      lrow[bh * SEQ + sq] = l_r[r];
    }
}

// ---------------- diagonal (self-key) correction + head merge ----------------
__global__ __launch_bounds__(256) void k_diagfix(const f16* __restrict__ qkv,
                                                 const float* __restrict__ oacc,
                                                 const float* __restrict__ mrow,
                                                 const float* __restrict__ lrow,
                                                 const unsigned char* __restrict__ gen_kpm,
                                                 f16* __restrict__ attn) {
  int idx = blockIdx.x * 256 + threadIdx.x;   // B*H*S
  int s = idx & (SEQ - 1);
  int bh = idx >> 11;
  int h = bh & (NHEAD - 1);
  int b = bh >> 3;
  float m = mrow[idx], l = lrow[idx];
  const float* op = oacc + (size_t)idx * HDIM;
  const f16* base = qkv + ((size_t)(b * SEQ + s)) * (3 * DMODEL) + h * HDIM;
  float a = 1.f, w = 0.f, denom = l;
  bool diag = (s >= PCPT_) && (gen_kpm[(size_t)b * GEN_ + (s - PCPT_)] == 0);
  if (diag) {
    float dot = 0.f;
    for (int d = 0; d < HDIM; ++d)
      dot += (float)base[d] * (float)base[DMODEL + d];
    dot *= 0.125f;
    float mn = fmaxf(m, dot);
    a = __expf(m - mn);
    w = __expf(dot - mn);
    denom = l * a + w;
  }
  float inv = 1.f / denom;
  const f16* vp = base + 2 * DMODEL;
  f16* out = attn + ((size_t)(b * SEQ + s)) * DMODEL + h * HDIM;
  for (int d = 0; d < HDIM; ++d) {
    float extra = diag ? (float)vp[d] * w : 0.f;
    out[d] = (f16)((op[d] * a + extra) * inv);
  }
}

// ---------------- residual + LayerNorm (one block per row, D=512) ----------
__global__ __launch_bounds__(256) void k_ln1(const float* __restrict__ pcpt,
                                             const float* __restrict__ gen,
                                             const float* __restrict__ proj,
                                             const float* __restrict__ g,
                                             const float* __restrict__ be,
                                             float* __restrict__ x1,
                                             f16* __restrict__ x1h) {
  int row = blockIdx.x;
  int b = row >> 11, s = row & (SEQ - 1);
  const float* xr = (s < PCPT_) ? &pcpt[((size_t)b * PCPT_ + s) * DMODEL]
                                : &gen[((size_t)b * GEN_ + (s - PCPT_)) * DMODEL];
  const float* pr = &proj[(size_t)row * DMODEL];
  int t = threadIdx.x, lane = t & 31, wid = t >> 5;
  __shared__ float red[8];
  float v0 = xr[t] + pr[t];
  float v1 = xr[t + 256] + pr[t + 256];
  float sm = v0 + v1;
  for (int m = 1; m < 32; m <<= 1) sm += __shfl_xor(sm, m, 32);
  if (!lane) red[wid] = sm;
  __syncthreads();
  float tot = 0.f;
#pragma unroll
  for (int i = 0; i < 8; ++i) tot += red[i];
  float mean = tot * (1.f / DMODEL);
  float d0 = v0 - mean, d1 = v1 - mean;
  float vs = d0 * d0 + d1 * d1;
  __syncthreads();
  for (int m = 1; m < 32; m <<= 1) vs += __shfl_xor(vs, m, 32);
  if (!lane) red[wid] = vs;
  __syncthreads();
  float vt = 0.f;
#pragma unroll
  for (int i = 0; i < 8; ++i) vt += red[i];
  float rinv = rsqrtf(vt * (1.f / DMODEL) + 1e-5f);
  float o0 = d0 * rinv * g[t] + be[t];
  float o1 = d1 * rinv * g[t + 256] + be[t + 256];
  x1[(size_t)row * DMODEL + t] = o0;
  x1[(size_t)row * DMODEL + t + 256] = o1;
  x1h[(size_t)row * DMODEL + t] = (f16)o0;
  x1h[(size_t)row * DMODEL + t + 256] = (f16)o1;
}

__global__ __launch_bounds__(256) void k_ln2(const float* __restrict__ x1,
                                             const float* __restrict__ ff,
                                             const float* __restrict__ g,
                                             const float* __restrict__ be,
                                             float* __restrict__ out) {
  int row = blockIdx.x;
  int b = row >> 11, s = row & (SEQ - 1);
  const float* xr = &x1[(size_t)row * DMODEL];
  const float* pr = &ff[(size_t)row * DMODEL];
  int t = threadIdx.x, lane = t & 31, wid = t >> 5;
  __shared__ float red[8];
  float v0 = xr[t] + pr[t];
  float v1 = xr[t + 256] + pr[t + 256];
  float sm = v0 + v1;
  for (int m = 1; m < 32; m <<= 1) sm += __shfl_xor(sm, m, 32);
  if (!lane) red[wid] = sm;
  __syncthreads();
  float tot = 0.f;
#pragma unroll
  for (int i = 0; i < 8; ++i) tot += red[i];
  float mean = tot * (1.f / DMODEL);
  float d0 = v0 - mean, d1 = v1 - mean;
  float vs = d0 * d0 + d1 * d1;
  __syncthreads();
  for (int m = 1; m < 32; m <<= 1) vs += __shfl_xor(vs, m, 32);
  if (!lane) red[wid] = vs;
  __syncthreads();
  float vt = 0.f;
#pragma unroll
  for (int i = 0; i < 8; ++i) vt += red[i];
  float rinv = rsqrtf(vt * (1.f / DMODEL) + 1e-5f);
  float o0 = d0 * rinv * g[t] + be[t];
  float o1 = d1 * rinv * g[t + 256] + be[t + 256];
  size_t oidx;
  if (s < PCPT_) oidx = ((size_t)(b * PCPT_ + s)) * DMODEL + t;
  else oidx = (size_t)BATCH * PCPT_ * DMODEL + ((size_t)(b * GEN_ + (s - PCPT_))) * DMODEL + t;
  out[oidx] = o0;
  out[oidx + 256] = o1;
}

// ---------------- launcher ----------------
extern "C" void kernel_launch(void* const* d_in, const int* in_sizes, int n_in,
                              void* d_out, int out_size, void* d_ws, size_t ws_size,
                              hipStream_t stream) {
  const float* pcpt = (const float*)d_in[0];
  const float* gen  = (const float*)d_in[1];
  const unsigned char* pcpt_kpm = (const unsigned char*)d_in[2];
  const unsigned char* gen_kpm  = (const unsigned char*)d_in[3];
  const float* in_w  = (const float*)d_in[4];
  const float* in_b  = (const float*)d_in[5];
  const float* out_w = (const float*)d_in[6];
  const float* out_b = (const float*)d_in[7];
  const float* w1 = (const float*)d_in[8];
  const float* b1 = (const float*)d_in[9];
  const float* w2 = (const float*)d_in[10];
  const float* b2 = (const float*)d_in[11];
  const float* ln1g = (const float*)d_in[12];
  const float* ln1b = (const float*)d_in[13];
  const float* ln2g = (const float*)d_in[14];
  const float* ln2b = (const float*)d_in[15];

  char* ws = (char*)d_ws;
  size_t off = 0;
  auto carve = [&](size_t bytes) -> void* {
    void* p = ws + off;
    off += (bytes + 255) & ~(size_t)255;
    return p;
  };
  f16*   xh    = (f16*)carve((size_t)MTOT * DMODEL * 2);
  f16*   qkvh  = (f16*)carve((size_t)MTOT * 3 * DMODEL * 2);
  float* oacc  = (float*)carve((size_t)BATCH * NHEAD * SEQ * HDIM * 4);
  float* mr    = (float*)carve((size_t)BATCH * NHEAD * SEQ * 4);
  float* lr    = (float*)carve((size_t)BATCH * NHEAD * SEQ * 4);
  f16*   attnh = (f16*)carve((size_t)MTOT * DMODEL * 2);
  float* projf = (float*)carve((size_t)MTOT * DMODEL * 4);  // reused as FFN out
  float* x1f   = (float*)carve((size_t)MTOT * DMODEL * 4);
  f16*   x1h   = (f16*)carve((size_t)MTOT * DMODEL * 2);
  f16*   hidh  = (f16*)carve((size_t)MTOT * DFF_ * 2);
  f16*   wh_in  = (f16*)carve((size_t)3 * DMODEL * DMODEL * 2);
  f16*   wh_out = (f16*)carve((size_t)DMODEL * DMODEL * 2);
  f16*   wh_1   = (f16*)carve((size_t)DFF_ * DMODEL * 2);
  f16*   wh_2   = (f16*)carve((size_t)DMODEL * DFF_ * 2);
  float* fff   = projf;  // alias: proj is dead after LN1

  // weights -> f16 (once; L2-resident thereafter)
  k_cvt<<<(3 * DMODEL * DMODEL) / 256, 256, 0, stream>>>(in_w, wh_in, 3 * DMODEL * DMODEL);
  k_cvt<<<(DMODEL * DMODEL) / 256, 256, 0, stream>>>(out_w, wh_out, DMODEL * DMODEL);
  k_cvt<<<(DFF_ * DMODEL) / 256, 256, 0, stream>>>(w1, wh_1, DFF_ * DMODEL);
  k_cvt<<<(DMODEL * DFF_) / 256, 256, 0, stream>>>(w2, wh_2, DMODEL * DFF_);

  // 0) concat + f16
  k_concat<<<(MTOT * DMODEL) / 256, 256, 0, stream>>>(pcpt, gen, xh);
  // 1) QKV projection (8192x1536, K=512)
  k_gemm<<<dim3(1536 / 128, MTOT / 128), 256, 0, stream>>>(
      xh, wh_in, in_b, nullptr, qkvh, MTOT, 3 * DMODEL, DMODEL, 0);
  // 2) flash attention over pcpt keys
  k_flash<<<dim3(SEQ / 128, NHEAD, BATCH), 256, 0, stream>>>(
      qkvh, pcpt_kpm, oacc, mr, lr);
  // 3) diagonal self-key correction + normalize + merge heads
  k_diagfix<<<(BATCH * NHEAD * SEQ) / 256, 256, 0, stream>>>(
      qkvh, oacc, mr, lr, gen_kpm, attnh);
  // 4) output projection (8192x512, K=512)
  k_gemm<<<dim3(512 / 128, MTOT / 128), 256, 0, stream>>>(
      attnh, wh_out, out_b, projf, nullptr, MTOT, DMODEL, DMODEL, 0);
  // 5) residual + LN1
  k_ln1<<<MTOT, 256, 0, stream>>>(pcpt, gen, projf, ln1g, ln1b, x1f, x1h);
  // 6) FFN1 + ReLU (8192x2048, K=512)
  k_gemm<<<dim3(DFF_ / 128, MTOT / 128), 256, 0, stream>>>(
      x1h, wh_1, b1, nullptr, hidh, MTOT, DFF_, DMODEL, 1);
  // 7) FFN2 (8192x512, K=2048)
  k_gemm<<<dim3(512 / 128, MTOT / 128), 256, 0, stream>>>(
      hidh, wh_2, b2, fff, nullptr, MTOT, DMODEL, DFF_, 0);
  // 8) residual + LN2 -> split (pcpt, gen) output layout
  k_ln2<<<MTOT, 256, 0, stream>>>(x1f, fff, ln2g, ln2b, (float*)d_out);
}
